// OlmoLMHeadCE_2826088481174
// MI455X (gfx1250) — matrix operand order repaired and checked by reading, not compiled, not run on hardware
//
#include <hip/hip_runtime.h>
#include <hip/hip_bf16.h>
#include <stdint.h>

// ---------------------------------------------------------------------------
// Fused LM-head + cross-entropy (sum reduction) for gfx1250 / MI455X.
//   losses_sum = sum_valid(lse[n] - logit[n, target[n]])
//   lse_sum    = sum_valid(lse[n])
// Strategy: bf16 WMMA (v_wmma_f32_16x16x32_bf16), W pre-transposed+converted
// to bf16 [V,H] (65.5MB, L2-resident), A staged in LDS per 128-row block,
// wave-local online logsumexp over V-slices, deterministic 2-stage reduce.
// ---------------------------------------------------------------------------

typedef __bf16 bf16_t;
typedef bf16_t v16bf __attribute__((ext_vector_type(16)));
typedef float  v8f   __attribute__((ext_vector_type(8)));

#define H_DIM   1024
#define V_DIM   32000
#define N_ROWS  4096
#define ROWS_PER_WG 128
#define WAVES   8
#define ASTRIDE 1032          // LDS row stride (elems): 1024 + 8 pad -> conflict-free b128
#define SLICES  8
#define NTILE64 (V_DIM / 64)  // 500 column tiles of 64
#define NBLK_FIN (N_ROWS / WAVES)  // 512 finalize blocks
#define IGNORE_INDEX (-100)

__device__ __forceinline__ uint16_t f2bf_rne(float f) {
  uint32_t u = __float_as_uint(f);
  uint32_t r = u + 0x7FFFu + ((u >> 16) & 1u);   // round-to-nearest-even
  return (uint16_t)(r >> 16);
}
__device__ __forceinline__ float bf2f(uint16_t h) {
  return __uint_as_float(((uint32_t)h) << 16);
}

union VecCast { uint4 q[2]; v16bf v; };
__device__ __forceinline__ v16bf make_frag(uint4 lo, uint4 hi) {
  VecCast u; u.q[0] = lo; u.q[1] = hi; return u.v;
}

// reductions across the 16-lane half-wave holding one matrix row
__device__ __forceinline__ float lane16_max(float v) {
  v = fmaxf(v, __shfl_xor(v, 1, 32));
  v = fmaxf(v, __shfl_xor(v, 2, 32));
  v = fmaxf(v, __shfl_xor(v, 4, 32));
  v = fmaxf(v, __shfl_xor(v, 8, 32));
  return v;
}
__device__ __forceinline__ float lane16_sum(float v) {
  v += __shfl_xor(v, 1, 32);
  v += __shfl_xor(v, 2, 32);
  v += __shfl_xor(v, 4, 32);
  v += __shfl_xor(v, 8, 32);
  return v;
}

// ---------------------------------------------------------------------------
// Kernel 1: W[H,V] f32 (row-major) -> whT[V,H] bf16 via LDS tile transpose.
// Coalesced on both sides; runs once, result lives in L2 (65.5MB < 192MB).
// ---------------------------------------------------------------------------
__global__ void __launch_bounds__(256)
wT_kernel(const float* __restrict__ w, uint16_t* __restrict__ whT) {
  __shared__ uint16_t tile[64][66];   // +2 pad: odd dword stride, conflict-free
  const int vt = blockIdx.x;          // 0..499  (V tiles of 64)
  const int kt = blockIdx.y;          // 0..15   (H tiles of 64)
  const int tx = threadIdx.x & 63;
  const int tq = threadIdx.x >> 6;    // 0..3
#pragma unroll
  for (int i = 0; i < 16; ++i) {
    int kr = tq * 16 + i;
    float f = w[(size_t)(kt * 64 + kr) * V_DIM + (size_t)vt * 64 + tx];
    tile[kr][tx] = f2bf_rne(f);
  }
  __syncthreads();
#pragma unroll
  for (int i = 0; i < 16; ++i) {
    int vr = tq * 16 + i;
    whT[(size_t)(vt * 64 + vr) * H_DIM + (size_t)kt * 64 + tx] = tile[tx][vr];
  }
}

// ---------------------------------------------------------------------------
// Kernel 2: main fused GEMM + online logsumexp.
// grid = 32 row-blocks x 8 V-slices; block = 256 (8 wave32).
// Each wave owns a 16-row stripe; all 8 waves walk the same column tiles so B
// fragments are shared through WGP$/L2.
// ---------------------------------------------------------------------------
__global__ void __launch_bounds__(256, 1)
lmce_main(const float* __restrict__ x, const uint16_t* __restrict__ whT,
          float* __restrict__ m_part, float* __restrict__ s_part) {
  extern __shared__ uint16_t smem[];          // [ROWS_PER_WG][ASTRIDE] bf16 A
  const int rowBlock = blockIdx.x & 31;       // N / 128 = 32
  const int slice    = blockIdx.x >> 5;       // 8 slices of V
  const int tid    = threadIdx.x;
  const int wave   = tid >> 5;
  const int lane   = tid & 31;
  const int laneLo = lane & 15;
  const int laneHi = lane >> 4;

  // ---- stage A block: 128 rows x 1024 K, f32 -> bf16 into padded LDS ----
  const float4* xblk = (const float4*)(x + (size_t)rowBlock * ROWS_PER_WG * H_DIM);
  for (int i = tid; i < ROWS_PER_WG * H_DIM / 4; i += 256) {
    float4 f = xblk[i];
    int e = i * 4;
    int r = e >> 10;            // / H
    int c = e & (H_DIM - 1);
    uint32_t p0 = (uint32_t)f2bf_rne(f.x) | ((uint32_t)f2bf_rne(f.y) << 16);
    uint32_t p1 = (uint32_t)f2bf_rne(f.z) | ((uint32_t)f2bf_rne(f.w) << 16);
    *(uint2*)(smem + (size_t)r * ASTRIDE + c) = make_uint2(p0, p1);
  }
  __syncthreads();

  // A-fragment base for this lane (ISA 16-bit A layout: lanes 0-15 hold K
  // chunks {k0..k0+7, k0+16..k0+23}; lanes 16-31 hold the +8 chunks).
  const uint16_t* aBase = smem + (size_t)(wave * 16 + laneLo) * ASTRIDE + laneHi * 8;

  float mrun[8], srun[8];
#pragma unroll
  for (int j = 0; j < 8; ++j) { mrun[j] = -3.0e38f; srun[j] = 0.0f; }

  for (int t64 = slice; t64 < NTILE64; t64 += SLICES) {
    const int cb = t64 * 64;
    // B layout in whT[V,H]: lane's fragment = K chunks {laneHi*8..+7, 16+laneHi*8..+7}
    // of column (cb+laneLo), matching the A fragment K order.
    const uint16_t* bBase = whT + (size_t)(cb + laneLo) * H_DIM + laneHi * 8;

    // speculative prefetch of the next tile's B stream (global_prefetch_b8)
    int tn = t64 + SLICES;
    if (tn < NTILE64)
      __builtin_prefetch(whT + (size_t)(tn * 64 + laneLo) * H_DIM, 0, 0);

    v8f acc0 = {}, acc1 = {}, acc2 = {}, acc3 = {};
#pragma unroll 4
    for (int k0 = 0; k0 < H_DIM; k0 += 32) {
      uint4 a0 = *(const uint4*)(aBase + k0);        // K = k0   + laneHi*8
      uint4 a1 = *(const uint4*)(aBase + k0 + 16);   // K = k0+16+ laneHi*8
      v16bf A = make_frag(a0, a1);
      const uint4* bp = (const uint4*)(bBase + k0);  // 16*H elems = 2048 uint4 per tile
      uint4 b00 = bp[0],          b01 = bp[2];
      uint4 b10 = bp[1 * 2048],   b11 = bp[1 * 2048 + 2];
      uint4 b20 = bp[2 * 2048],   b21 = bp[2 * 2048 + 2];
      uint4 b30 = bp[3 * 2048],   b31 = bp[3 * 2048 + 2];
      acc0 = __builtin_amdgcn_wmma_f32_16x16x32_bf16(false, A, false, make_frag(b00, b01),
                                                     (short)0, acc0, false, false);
      acc1 = __builtin_amdgcn_wmma_f32_16x16x32_bf16(false, A, false, make_frag(b10, b11),
                                                     (short)0, acc1, false, false);
      acc2 = __builtin_amdgcn_wmma_f32_16x16x32_bf16(false, A, false, make_frag(b20, b21),
                                                     (short)0, acc2, false, false);
      acc3 = __builtin_amdgcn_wmma_f32_16x16x32_bf16(false, A, false, make_frag(b30, b31),
                                                     (short)0, acc3, false, false);
    }

    // ---- online logsumexp update over these 64 columns ----
    // C/D layout: VGPR j holds row j (lanes 0-15) and row j+8 (lanes 16-31).
#pragma unroll
    for (int j = 0; j < 8; ++j) {
      float tmax = fmaxf(fmaxf(acc0[j], acc1[j]), fmaxf(acc2[j], acc3[j]));
      tmax = lane16_max(tmax);
      float mnew  = fmaxf(mrun[j], tmax);
      float scale = __expf(mrun[j] - mnew);
      float es = __expf(acc0[j] - mnew) + __expf(acc1[j] - mnew)
               + __expf(acc2[j] - mnew) + __expf(acc3[j] - mnew);
      es = lane16_sum(es);
      srun[j] = srun[j] * scale + es;
      mrun[j] = mnew;
    }
  }

  // write per-slice partials: row of VGPR j is j + laneHi*8
  if (laneLo == 0) {
#pragma unroll
    for (int j = 0; j < 8; ++j) {
      int row = rowBlock * ROWS_PER_WG + wave * 16 + j + laneHi * 8;
      m_part[(size_t)row * SLICES + slice] = mrun[j];
      s_part[(size_t)row * SLICES + slice] = srun[j];
    }
  }
}

// ---------------------------------------------------------------------------
// Kernel 3: per-row finalize. One wave per row: target-logit dot (bf16 to
// match GEMM numerics), merge 8 slice partials into lse, per-block partial
// sums (deterministic — no float atomics).
// ---------------------------------------------------------------------------
__global__ void __launch_bounds__(256)
lmce_finalize(const float* __restrict__ x, const uint16_t* __restrict__ whT,
              const int* __restrict__ target,
              const float* __restrict__ m_part, const float* __restrict__ s_part,
              float* __restrict__ blockPart) {
  __shared__ float redA[WAVES], redB[WAVES];
  const int wave = threadIdx.x >> 5;
  const int lane = threadIdx.x & 31;
  const int n = blockIdx.x * WAVES + wave;     // 512 * 8 = 4096 rows

  int  t = target[n];
  bool valid = (t != IGNORE_INDEX);
  int  ts = valid ? t : 0;
  const float*    xr = x   + (size_t)n  * H_DIM;
  const uint16_t* wr = whT + (size_t)ts * H_DIM;
  float acc = 0.0f;
  for (int k = lane; k < H_DIM; k += 32)
    acc = fmaf(bf2f(f2bf_rne(xr[k])), bf2f(wr[k]), acc);
#pragma unroll
  for (int m = 16; m >= 1; m >>= 1) acc += __shfl_xor(acc, m, 32);

  if (lane == 0) {
    float M = -3.0e38f;
#pragma unroll
    for (int s = 0; s < SLICES; ++s) M = fmaxf(M, m_part[(size_t)n * SLICES + s]);
    float S = 0.0f;
#pragma unroll
    for (int s = 0; s < SLICES; ++s)
      S += s_part[(size_t)n * SLICES + s] * __expf(m_part[(size_t)n * SLICES + s] - M);
    float lse = M + __logf(S);
    redA[wave] = valid ? (lse - acc) : 0.0f;
    redB[wave] = valid ? lse : 0.0f;
  }
  __syncthreads();
  if (threadIdx.x == 0) {
    float a = 0.0f, b = 0.0f;
#pragma unroll
    for (int w = 0; w < WAVES; ++w) { a += redA[w]; b += redB[w]; }
    blockPart[blockIdx.x]            = a;
    blockPart[NBLK_FIN + blockIdx.x] = b;
  }
}

// Kernel 4: final deterministic reduction of 512 block partials -> 2 scalars.
__global__ void __launch_bounds__(256)
lmce_reduce(const float* __restrict__ blockPart, float* __restrict__ out) {
  __shared__ float sa[256], sb[256];
  float a = 0.0f, b = 0.0f;
  for (int i = threadIdx.x; i < NBLK_FIN; i += 256) {
    a += blockPart[i];
    b += blockPart[NBLK_FIN + i];
  }
  sa[threadIdx.x] = a; sb[threadIdx.x] = b;
  __syncthreads();
  if (threadIdx.x == 0) {
    float ta = 0.0f, tb = 0.0f;
    for (int i = 0; i < 256; ++i) { ta += sa[i]; tb += sb[i]; }
    out[0] = ta;   // losses.sum()
    out[1] = tb;   // lse_masked.sum()
  }
}

// ---------------------------------------------------------------------------
extern "C" void kernel_launch(void* const* d_in, const int* in_sizes, int n_in,
                              void* d_out, int out_size, void* d_ws, size_t ws_size,
                              hipStream_t stream) {
  (void)in_sizes; (void)n_in; (void)out_size; (void)ws_size;
  const float* x      = (const float*)d_in[0];
  const float* w      = (const float*)d_in[1];
  const int*   target = (const int*)d_in[2];
  float* out = (float*)d_out;

  char* ws = (char*)d_ws;
  uint16_t* whT = (uint16_t*)ws;                       // 65,536,000 B
  size_t off = (size_t)V_DIM * H_DIM * sizeof(uint16_t);
  float* m_part = (float*)(ws + off); off += (size_t)N_ROWS * SLICES * sizeof(float);
  float* s_part = (float*)(ws + off); off += (size_t)N_ROWS * SLICES * sizeof(float);
  float* blockPart = (float*)(ws + off);               // 2*512 floats

  // 1) weight -> bf16 [V,H] (L2-resident working set for the GEMM)
  wT_kernel<<<dim3(V_DIM / 64, H_DIM / 64), 256, 0, stream>>>(w, whT);

  // 2) fused GEMM + online logsumexp (dynamic LDS: 128 x 1032 bf16 ~ 258KB)
  size_t smem = (size_t)ROWS_PER_WG * ASTRIDE * sizeof(uint16_t);
  lmce_main<<<dim3((N_ROWS / ROWS_PER_WG) * SLICES), 256, smem, stream>>>(
      x, whT, m_part, s_part);

  // 3) per-row finalize + 4) deterministic reduction
  lmce_finalize<<<dim3(NBLK_FIN), 256, 0, stream>>>(x, whT, target, m_part, s_part, blockPart);
  lmce_reduce<<<1, 256, 0, stream>>>(blockPart, out);
}
